// RGCN_65609920414432
// MI455X (gfx1250) — matrix operand order repaired and checked by reading, not compiled, not run on hardware
//
#include <hip/hip_runtime.h>
#include <hip/hip_bf16.h>

typedef __attribute__((ext_vector_type(16))) __bf16 v16bf;
typedef __attribute__((ext_vector_type(8)))  __bf16 v8bf;
typedef __attribute__((ext_vector_type(4)))  __bf16 v4bf;
typedef __attribute__((ext_vector_type(8)))  float  v8f;

#define D_DIM 128
#define R_REL 8
#define K_TOT (R_REL * D_DIM + D_DIM)   // 1152
#define K_TILES (K_TOT / 32)            // 36
#define BTILE   (8 * 32 * 16)           // bf16 elems per B K-tile (8KB)
#define SCAN_T 256
#define SCAN_I 4                        // 1024 items per scan block

// ---------------- generic zero (16B granularity) ----------------
__global__ __launch_bounds__(256) void zero4_kernel(float4* __restrict__ p, long n4) {
  long i = (long)blockIdx.x * blockDim.x + threadIdx.x;
  if (i < n4) p[i] = make_float4(0.f, 0.f, 0.f, 0.f);
}

// ---------------- fp32 -> bf16 convert (4 elems/thread) ----------------
__global__ __launch_bounds__(256) void cvt_bf16_kernel(
    const float* __restrict__ x, __bf16* __restrict__ xb, long n4)
{
  long i = (long)blockIdx.x * blockDim.x + threadIdx.x;
  if (i >= n4) return;
  const float4 v = *(const float4*)(x + i * 4);
  v4bf o;
  o[0] = (__bf16)v.x; o[1] = (__bf16)v.y; o[2] = (__bf16)v.z; o[3] = (__bf16)v.w;
  *(v4bf*)(xb + i * 4) = o;
}

// ---------------- CSR build: degree histogram ----------------
__global__ __launch_bounds__(256) void deg_kernel(
    const int* __restrict__ dst, const int* __restrict__ et,
    int* __restrict__ deg, int E)
{
  int e = blockIdx.x * blockDim.x + threadIdx.x;
  if (e >= E) return;
  atomicAdd(deg + (size_t)dst[e] * R_REL + et[e], 1);
}

// ---------------- scan pass 1: per-block exclusive scan + block sums ----------------
__global__ __launch_bounds__(SCAN_T) void scan1_kernel(
    const int* __restrict__ deg, int* __restrict__ offs,
    int* __restrict__ bsums, int n)
{
  __shared__ int sd[SCAN_T];
  int t = threadIdx.x;
  long base = (long)blockIdx.x * SCAN_T * SCAN_I + (long)t * SCAN_I;
  int v[SCAN_I];
  int tsum = 0;
#pragma unroll
  for (int i = 0; i < SCAN_I; ++i) {
    long idx = base + i;
    v[i] = (idx < n) ? deg[idx] : 0;
    tsum += v[i];
  }
  sd[t] = tsum;
  __syncthreads();
  for (int off = 1; off < SCAN_T; off <<= 1) {
    int a = (t >= off) ? sd[t - off] : 0;
    __syncthreads();
    sd[t] += a;
    __syncthreads();
  }
  if (t == SCAN_T - 1) bsums[blockIdx.x] = sd[t];
  int acc = sd[t] - tsum;   // exclusive base for this thread
#pragma unroll
  for (int i = 0; i < SCAN_I; ++i) {
    long idx = base + i;
    if (idx < n) offs[idx] = acc;
    acc += v[i];
  }
}

// ---------------- scan pass 2: exclusive scan of block sums (tiny, serial) ----------------
__global__ void scan2_kernel(int* __restrict__ bsums, int nb) {
  if (threadIdx.x == 0 && blockIdx.x == 0) {
    int acc = 0;
    for (int i = 0; i < nb; ++i) { int v = bsums[i]; bsums[i] = acc; acc += v; }
  }
}

// ---------------- scan pass 3: add block offsets, init fill cursors ----------------
__global__ __launch_bounds__(256) void scan3_kernel(
    int* __restrict__ offs, const int* __restrict__ bsums,
    int* __restrict__ pos, int n)
{
  int i = blockIdx.x * blockDim.x + threadIdx.x;
  if (i >= n) return;
  int o = offs[i] + bsums[i / (SCAN_T * SCAN_I)];
  offs[i] = o;
  pos[i] = o;
}

// ---------------- CSR fill: per-segment source-node lists ----------------
__global__ __launch_bounds__(256) void fill_kernel(
    const int* __restrict__ src, const int* __restrict__ dst,
    const int* __restrict__ et, int* __restrict__ pos,
    int* __restrict__ srcs, int E)
{
  int e = blockIdx.x * blockDim.x + threadIdx.x;
  if (e >= E) return;
  int s = dst[e] * R_REL + et[e];
  int idx = atomicAdd(pos + s, 1);
  srcs[idx] = src[e];
}

// ---------------- per-segment mean: bf16 gather, fp32 accumulate, bf16 out ----------------
__global__ __launch_bounds__(256) void mean_kernel(
    const int* __restrict__ offs, const int* __restrict__ deg,
    const int* __restrict__ srcs, const __bf16* __restrict__ xb,
    __bf16* __restrict__ meanb, int NR)
{
  long gid = (long)blockIdx.x * blockDim.x + threadIdx.x;
  int s = (int)(gid >> 5);
  int t = (int)(gid & 31);
  if (s >= NR) return;
  int st = offs[s];
  int dg = deg[s];
  float a0 = 0.f, a1 = 0.f, a2 = 0.f, a3 = 0.f;
  for (int j = 0; j < dg; ++j) {
    int sn = srcs[st + j];
    const v4bf v = *(const v4bf*)(xb + (size_t)sn * D_DIM + t * 4);
    a0 += (float)v[0]; a1 += (float)v[1]; a2 += (float)v[2]; a3 += (float)v[3];
  }
  float sc = 1.0f / (float)(dg > 0 ? dg : 1);
  v4bf o;
  o[0] = (__bf16)(a0 * sc); o[1] = (__bf16)(a1 * sc);
  o[2] = (__bf16)(a2 * sc); o[3] = (__bf16)(a3 * sc);
  *(v4bf*)(meanb + (size_t)s * D_DIM + t * 4) = o;
}

// ---------------- pack Wc = [W_0..W_7 ; root] into per-lane WMMA B fragments ----------------
__global__ __launch_bounds__(256) void pack_b_kernel(
    const float* __restrict__ W, const float* __restrict__ root,
    __bf16* __restrict__ Bp)
{
  int gid = blockIdx.x * 256 + threadIdx.x;
  if (gid >= K_TILES * BTILE) return;
  int idx  = gid & 15;
  int lane = (gid >> 4) & 31;
  int ct   = (gid >> 9) & 7;
  int kt   = gid >> 12;
  int n    = ct * 16 + (lane & 15);
  int hlf  = lane >> 4;
  int kl   = (idx < 8) ? (hlf * 8 + idx) : (16 + hlf * 8 + (idx - 8));   // B 32x16 bf16: lanes 0-15 K=0..7,16..23; lanes 16-31 K=8..15,24..31
  int k    = kt * 32 + kl;
  float v;
  if (k < R_REL * D_DIM)
    v = W[((size_t)(k >> 7) * D_DIM + (k & 127)) * D_DIM + n];   // W[r][d][n]
  else
    v = root[(size_t)(k - R_REL * D_DIM) * D_DIM + n];
  Bp[gid] = (__bf16)v;
}

// ---------------- fused layer GEMM: out = relu([mean | x] @ [W;root] + bias) ----------------
// B K-tiles staged through LDS (double-buffered, 16KB); A is two b128 loads/K-step.
template <bool OUT_BF16>
__global__ __launch_bounds__(256) void rgcn_gemm_kernel(
    const __bf16* __restrict__ meanb, const __bf16* __restrict__ xroot,
    const __bf16* __restrict__ Bp, const float* __restrict__ bias,
    float* __restrict__ outf, __bf16* __restrict__ outb, int nTiles)
{
  __shared__ __align__(32) __bf16 ldsB[2][BTILE];

  int tid  = threadIdx.x;
  int lane = tid & 31;
  int wave = tid >> 5;
  int tile = blockIdx.x * 8 + wave;       // 16 output rows per wave
  bool active = tile < nTiles;            // wave-uniform; no early return (barriers below)
  int m   = lane & 15;
  int hlf = lane >> 4;
  int row = tile * 16 + m;

  v8f acc[8] = {};                        // 8 col-tiles x (16x16 f32)

  // preload B K-tile 0 into LDS (each thread: 32B slice)
  {
    v16bf st = *(const v16bf*)(Bp + (size_t)tid * 16);
    *(v16bf*)&ldsB[0][tid * 16] = st;
  }
  __syncthreads();

  for (int kt = 0; kt < K_TILES; ++kt) {
    int cur = kt & 1;
    bool hasNext = (kt + 1) < K_TILES;
    v16bf nx;
    if (hasNext)                          // prefetch next B K-tile to regs
      nx = *(const v16bf*)(Bp + (size_t)(kt + 1) * BTILE + tid * 16);

    if (active) {
      // A fragment: 16x32 bf16, per-lane K chunks {h*8..+7} U {16+h*8..+7}
      const __bf16* base = (kt < 32)
          ? (meanb + (size_t)row * (R_REL * D_DIM) + kt * 32)   // mean part [N][1024]
          : (xroot + (size_t)row * D_DIM + (kt - 32) * 32);     // root part
      v8bf lo = *(const v8bf*)(base + hlf * 8);
      v8bf hi = *(const v8bf*)(base + 16 + hlf * 8);
      v16bf a = __builtin_shufflevector(lo, hi,
          0, 1, 2, 3, 4, 5, 6, 7, 8, 9, 10, 11, 12, 13, 14, 15);

      v16bf bfrag[8];
#pragma unroll
      for (int ctile = 0; ctile < 8; ++ctile)
        bfrag[ctile] = *(const v16bf*)&ldsB[cur][(ctile * 32 + lane) * 16];
#pragma unroll
      for (int ctile = 0; ctile < 8; ++ctile)
        acc[ctile] = __builtin_amdgcn_wmma_f32_16x16x32_bf16(
            false, a, false, bfrag[ctile], (short)0, acc[ctile], false, false);
    }

    __syncthreads();                      // all waves done reading ldsB[1-cur] (prev iter)
    if (hasNext)
      *(v16bf*)&ldsB[1 - cur][tid * 16] = nx;
    __syncthreads();                      // staged tile visible to all waves
  }

  if (active) {
    // epilogue: C/D layout -> lane m = col, VGPR j = row j (+8 for hi lanes)
#pragma unroll
    for (int ctile = 0; ctile < 8; ++ctile) {
      int col = ctile * 16 + m;
      float bv = bias[col];
#pragma unroll
      for (int j = 0; j < 8; ++j) {
        int orow = tile * 16 + hlf * 8 + j;
        float v = acc[ctile][j] + bv;
        v = v > 0.f ? v : 0.f;
        if (OUT_BF16)
          outb[(size_t)orow * D_DIM + col] = (__bf16)v;
        else
          outf[(size_t)orow * D_DIM + col] = v;
      }
    }
  }
}

__global__ __launch_bounds__(256) void copy_kernel(const float* __restrict__ s,
                                                   float* __restrict__ d, int n) {
  int i = blockIdx.x * blockDim.x + threadIdx.x;
  if (i < n) d[i] = s[i];
}

extern "C" void kernel_launch(void* const* d_in, const int* in_sizes, int n_in,
                              void* d_out, int out_size, void* d_ws, size_t ws_size,
                              hipStream_t stream)
{
  const float* x     = (const float*)d_in[0];
  const int*   ei    = (const int*)d_in[1];
  const int*   et    = (const int*)d_in[2];
  const float* W1    = (const float*)d_in[3];
  const float* root1 = (const float*)d_in[4];
  const float* b1    = (const float*)d_in[5];
  const float* W2    = (const float*)d_in[6];
  const float* root2 = (const float*)d_in[7];
  const float* b2    = (const float*)d_in[8];
  const float* rel   = (const float*)d_in[9];

  const int N  = in_sizes[0] / D_DIM;   // 50000 (multiple of 16)
  const int E  = in_sizes[2];           // 600000
  const int NR = N * R_REL;             // 400000 segments
  const int* srcp = ei;
  const int* dstp = ei + E;

  // workspace layout (bf16 data path)
  __bf16* meanb = (__bf16*)d_ws;                              // N*R*D bf16 (102.4MB)
  __bf16* xb    = meanb + (size_t)N * R_REL * D_DIM;          // N*D bf16 (x as bf16)
  __bf16* h1b   = xb + (size_t)N * D_DIM;                     // N*D bf16 (layer-1 out)
  __bf16* Bp    = h1b + (size_t)N * D_DIM;                    // 36*8*32*16 bf16 (288KB)
  int*    deg   = (int*)(Bp + (size_t)K_TILES * BTILE);
  int*    offs  = deg  + NR;
  int*    pos   = offs + NR;
  int*    srcs  = pos  + NR;                                  // E ints
  int*    bsums = srcs + E;

  float* outh = (float*)d_out;

  const int nScanBlocks = (NR + SCAN_T * SCAN_I - 1) / (SCAN_T * SCAN_I);  // 391
  const int egrid  = (E + 255) / 256;
  const int pgrid  = (K_TILES * BTILE + 255) / 256;
  const int nTiles = N / 16;
  const int ggrid  = (nTiles + 7) / 8;
  const long mthreads = (long)NR * 32;
  const int mgrid  = (int)((mthreads + 255) / 256);
  const long xcvt4 = (long)N * D_DIM / 4;

  // ---- CSR build by segment (dst*R + et), once per call ----
  zero4_kernel<<<(NR / 4 + 255) / 256, 256, 0, stream>>>((float4*)deg, NR / 4);
  deg_kernel<<<egrid, 256, 0, stream>>>(dstp, et, deg, E);
  scan1_kernel<<<nScanBlocks, SCAN_T, 0, stream>>>(deg, offs, bsums, NR);
  scan2_kernel<<<1, 64, 0, stream>>>(bsums, nScanBlocks);
  scan3_kernel<<<(NR + 255) / 256, 256, 0, stream>>>(offs, bsums, pos, NR);
  fill_kernel<<<egrid, 256, 0, stream>>>(srcp, dstp, et, pos, srcs, E);

  // ---- x -> bf16 once ----
  cvt_bf16_kernel<<<(int)((xcvt4 + 255) / 256), 256, 0, stream>>>(x, xb, xcvt4);

  // ---- layer 1 (bf16 hidden output) ----
  mean_kernel<<<mgrid, 256, 0, stream>>>(offs, deg, srcs, xb, meanb, NR);
  pack_b_kernel<<<pgrid, 256, 0, stream>>>(W1, root1, Bp);
  rgcn_gemm_kernel<true><<<ggrid, 256, 0, stream>>>(meanb, xb, Bp, b1, nullptr, h1b, nTiles);

  // ---- layer 2 (fp32 final output) ----
  mean_kernel<<<mgrid, 256, 0, stream>>>(offs, deg, srcs, h1b, meanb, NR);
  pack_b_kernel<<<pgrid, 256, 0, stream>>>(W2, root2, Bp);
  rgcn_gemm_kernel<false><<<ggrid, 256, 0, stream>>>(meanb, h1b, Bp, b2, outh, nullptr, nTiles);

  // ---- rel_emb passthrough (tuple output) ----
  copy_kernel<<<(R_REL * D_DIM + 255) / 256, 256, 0, stream>>>(
      rel, outh + (size_t)N * D_DIM, R_REL * D_DIM);
}
